// TransformerXL_57552561766563
// MI455X (gfx1250) — matrix operand order repaired and checked by reading, not compiled, not run on hardware
//
#include <hip/hip_runtime.h>

// ---------------- model dims ----------------
#define QL   1024
#define KLN  2048
#define DM   512
#define NH   8
#define DHD  64
#define DI   2048
#define NL   6
#define SCALE_ATTN 0.125f   // 1/sqrt(64)

typedef __attribute__((ext_vector_type(16))) _Float16 v16h;
typedef __attribute__((ext_vector_type(8)))  _Float16 h8;
typedef __attribute__((ext_vector_type(8)))  float    v8f;

// ---------------------------------------------------------------------
// CDNA5 async global->LDS copy (ASYNCcnt-tracked), per CDNA5 ISA.
// VDST carries the per-lane LDS byte address; generic shared pointers on
// amdgcn have the LDS offset in their low 32 bits, so truncation == the
// addrspacecast result.
// ---------------------------------------------------------------------
__device__ __forceinline__ void async_copy_b128(const _Float16* g, _Float16* l) {
  unsigned          lds = (unsigned)(unsigned long long)l;
  unsigned long long ga = (unsigned long long)g;
  asm volatile("global_load_async_to_lds_b128 %0, %1, off"
               :: "v"(lds), "v"(ga) : "memory");
}

__device__ __forceinline__ void wait_async0() {
#if __has_builtin(__builtin_amdgcn_s_wait_asynccnt)
  __builtin_amdgcn_s_wait_asynccnt(0);
#else
  asm volatile("s_wait_asynccnt 0x0" ::: "memory");
#endif
}

// =====================================================================
// WMMA GEMM:  C(M,N) = alpha * A(M,K) x B + bias, optional ReLU
//   A: f16 row-major, lda (K-contiguous).  Requires M % 128 == 0 (true at
//      every call site: M is 1024 or 2048) -> A staging is unguarded async.
//   B: f16. transB=1: element (k,n) at B[n*ldb + k] (x @ W.T, W row-major);
//            requires N % 128 == 0 (true at every transB=1 call site) ->
//            unguarded async staging.
//           transB=0: element (k,n) at B[k*ldb + n]; guarded scalar staging
//            (used by prob x V, N = 64).
//   C: f32 row-major, ldc.
// Block: 256 threads = 8 wave32 in a 4(M) x 2(N) grid; block tile 128x128,
// K-step 32. Each wave owns a 32x64 tile = 2x4 grid of 16x16 WMMA tiles
// (8 x v_wmma_f32_16x16x32_f16 per K-step from 2 A-frags + 4 B-frags).
// =====================================================================
#define BM 128
#define BN 128
#define BK 32

__global__ __launch_bounds__(256)
void gemm_wmma_f16(const _Float16* __restrict__ A, int lda,
                   const _Float16* __restrict__ B, int ldb, int transB,
                   float* __restrict__ C, int ldc,
                   const float* __restrict__ bias,
                   int M, int N, int K, float alpha, int relu)
{
  __shared__ _Float16 As[BM][BK + 8];   // +8 pad keeps 16B alignment, avoids conflicts
  __shared__ _Float16 Bs[BN][BK + 8];   // stored [n][k]

  const int tid  = threadIdx.x;
  const int lane = tid & 31;
  const int wave = tid >> 5;          // 0..7
  const int wm   = wave >> 1;         // 0..3 -> 32 rows each
  const int wn   = wave & 1;          // 0..1 -> 64 cols each
  const int m0   = blockIdx.y * BM;
  const int n0   = blockIdx.x * BN;

  v8f zero = {};
  v8f acc[2][4];
  #pragma unroll
  for (int i = 0; i < 2; ++i)
    #pragma unroll
    for (int j = 0; j < 4; ++j) acc[i][j] = zero;

  const int r  = tid >> 1;            // 0..127
  const int kc = (tid & 1) * 16;      // 0 or 16

  for (int k0 = 0; k0 < K; k0 += BK) {
    // ---- stage A tile 128x32 via async global->LDS (2 x b128 per thread) ----
    {
      const _Float16* pa = A + (size_t)(m0 + r) * lda + k0 + kc;
      async_copy_b128(pa,     &As[r][kc]);
      async_copy_b128(pa + 8, &As[r][kc + 8]);
    }
    // ---- stage B tile as Bs[n][k] ----
    if (transB) {
      const _Float16* pb = B + (size_t)(n0 + r) * ldb + k0 + kc;
      async_copy_b128(pb,     &Bs[r][kc]);
      async_copy_b128(pb + 8, &Bs[r][kc + 8]);
    } else {
      const int gn = n0 + r;
      #pragma unroll
      for (int kk = 0; kk < 16; ++kk) {
        _Float16 val = (_Float16)0.f;
        if (gn < N) val = B[(size_t)(k0 + kc + kk) * ldb + gn];
        Bs[r][kc + kk] = val;
      }
      if (gn < N && k0 + BK < K)
        __builtin_prefetch(B + (size_t)(k0 + BK) * ldb + gn, 0, 1);
    }
    wait_async0();
    __syncthreads();

    // ---- fragments per ISA 16-bit A(16x32)/B(32x16) lane layouts ----
    v16h af[2], bf[4];
    {
      const int mrow = lane & 15;
      const int koff = (lane >> 4) * 8;   // lanes 0-15: K 0..7/16..23 ; 16-31: 8..15/24..31
      #pragma unroll
      for (int i = 0; i < 2; ++i) {
        const int m = wm * 32 + i * 16 + mrow;
        h8 lo = *(const h8*)&As[m][koff];
        h8 hi = *(const h8*)&As[m][koff + 16];
        #pragma unroll
        for (int e = 0; e < 8; ++e) { af[i][e] = lo[e]; af[i][e + 8] = hi[e]; }
      }
      const int kb = (lane >> 4) * 8;     // lanes 0-15: K 0..7/16..23 ; 16-31: 8..15/24..31
      #pragma unroll
      for (int j = 0; j < 4; ++j) {
        const int n = wn * 64 + j * 16 + mrow;
        h8 lo = *(const h8*)&Bs[n][kb];
        h8 hi = *(const h8*)&Bs[n][kb + 16];
        #pragma unroll
        for (int e = 0; e < 8; ++e) { bf[j][e] = lo[e]; bf[j][e + 8] = hi[e]; }
      }
    }

    #pragma unroll
    for (int i = 0; i < 2; ++i)
      #pragma unroll
      for (int j = 0; j < 4; ++j)
        acc[i][j] = __builtin_amdgcn_wmma_f32_16x16x32_f16(
            false, af[i], false, bf[j], (short)0, acc[i][j], false, false);

    __syncthreads();
  }

  // ---- epilogue: C/D layout -> VGPR v: lanes0-15 M=v, lanes16-31 M=8+v ----
  const int ncol  = lane & 15;
  const int mhalf = (lane >> 4) * 8;
  #pragma unroll
  for (int i = 0; i < 2; ++i) {
    #pragma unroll
    for (int j = 0; j < 4; ++j) {
      const int gn  = n0 + wn * 64 + j * 16 + ncol;
      const int gmb = m0 + wm * 32 + i * 16 + mhalf;
      const float bv = (bias != nullptr && gn < N) ? bias[gn] : 0.f;
      #pragma unroll
      for (int v = 0; v < 8; ++v) {
        const int gm = gmb + v;
        if (gm < M && gn < N) {
          float val = alpha * acc[i][j][v] + bv;
          if (relu) val = val > 0.f ? val : 0.f;
          C[(size_t)gm * ldc + gn] = val;
        }
      }
    }
  }
}

// =====================================================================
// Support kernels
// =====================================================================
__global__ void cvt_f32_to_f16(const float* __restrict__ s, _Float16* __restrict__ d, int n) {
  for (int i = blockIdx.x * blockDim.x + threadIdx.x; i < n; i += gridDim.x * blockDim.x)
    d[i] = (_Float16)s[i];
}

__global__ void copy_f32(float* __restrict__ d, const float* __restrict__ s, int n) {
  for (int i = blockIdx.x * blockDim.x + threadIdx.x; i < n; i += gridDim.x * blockDim.x)
    d[i] = s[i];
}

// u[q,c] = sum_t entr_w[t] * dec_inp[0,t,q,c]
__global__ void track_combine(const float* __restrict__ dec, const float* __restrict__ tw,
                              float* __restrict__ u) {
  int i = blockIdx.x * blockDim.x + threadIdx.x;
  if (i >= QL * 128) return;
  float acc = 0.f;
  #pragma unroll
  for (int t = 0; t < 4; ++t) acc += tw[t] * dec[t * QL * 128 + i];
  u[i] = acc;
}

// bias1[d] = (sum_t entr_w[t]) * enp1_b[d] + entr_b
__global__ void make_bias1(const float* __restrict__ enp1_b, const float* __restrict__ tw,
                           const float* __restrict__ tb, float* __restrict__ bias1) {
  int d = blockIdx.x * blockDim.x + threadIdx.x;
  if (d >= DM) return;
  float sw = tw[0] + tw[1] + tw[2] + tw[3];
  bias1[d] = enp1_b[d] * sw + tb[0];
}

// pos_emb[j,d] : pos = KLN-1-j ; d<256 -> sin(pos*invf[d]) ; else cos(pos*invf[d-256])
__global__ void pos_emb_make(float* __restrict__ pos) {
  int i = blockIdx.x * blockDim.x + threadIdx.x;
  if (i >= KLN * DM) return;
  int j = i >> 9, d = i & (DM - 1);
  float p = (float)(KLN - 1 - j);
  float freq = __expf(-9.210340371976184f * (float)(d & 255) * (1.f / 256.f)); // 10000^-x
  float a = p * freq;
  pos[i] = (d < 256) ? __sinf(a) : __cosf(a);
}

// cat16[j,d] = f16( j<QL ? mem[j,d] : core[j-QL,d] )
__global__ void concat_cat16(const float* __restrict__ mem, const float* __restrict__ core,
                             _Float16* __restrict__ cat16) {
  int i = blockIdx.x * blockDim.x + threadIdx.x;
  if (i >= KLN * DM) return;
  int row = i >> 9, d = i & (DM - 1);
  float v = (row < QL) ? mem[(size_t)row * DM + d] : core[(size_t)(row - QL) * DM + d];
  cat16[i] = (_Float16)v;
}

// split heads (KLN x 1536) -> qw/qr (QL x 512, +rel biases), k16/v16 (KLN x 512)
__global__ void split_qkv(const float* __restrict__ heads,
                          const float* __restrict__ rwb, const float* __restrict__ rrb,
                          _Float16* __restrict__ qw, _Float16* __restrict__ qr,
                          _Float16* __restrict__ k16, _Float16* __restrict__ v16) {
  int i = blockIdx.x * blockDim.x + threadIdx.x;
  if (i >= KLN * 3 * DM) return;
  int row = i / (3 * DM), c = i % (3 * DM);
  float h = heads[i];
  if (c < DM) {
    if (row >= (KLN - QL)) {
      int q = row - (KLN - QL);
      qw[(size_t)q * DM + c] = (_Float16)(h + rwb[c]);
      qr[(size_t)q * DM + c] = (_Float16)(h + rrb[c]);
    }
  } else if (c < 2 * DM) {
    k16[(size_t)row * DM + (c - DM)] = (_Float16)h;
  } else {
    v16[(size_t)row * DM + (c - 2 * DM)] = (_Float16)h;
  }
}

// fused rel-shift + mask + softmax over j:
//   s(i,j) = (AC[i,j] + BDraw[i, j+1023-i]) * SCALE  for j <= 1024+i, else masked
__global__ __launch_bounds__(256)
void softmax_rel(const float* __restrict__ AC, const float* __restrict__ BD,
                 _Float16* __restrict__ prob) {
  __shared__ float red[256];
  const int i = blockIdx.x, tid = threadIdx.x;
  const int jmax = (KLN - QL) + i;              // inclusive
  const size_t rowA = (size_t)i * KLN;
  float mx = -3.4e38f;
  for (int j = tid; j < KLN; j += 256)
    if (j <= jmax) {
      float s = (AC[rowA + j] + BD[rowA + j + (QL - 1) - i]) * SCALE_ATTN;
      mx = fmaxf(mx, s);
    }
  red[tid] = mx; __syncthreads();
  for (int s = 128; s > 0; s >>= 1) { if (tid < s) red[tid] = fmaxf(red[tid], red[tid + s]); __syncthreads(); }
  mx = red[0]; __syncthreads();
  float sum = 0.f;
  for (int j = tid; j < KLN; j += 256)
    if (j <= jmax)
      sum += __expf((AC[rowA + j] + BD[rowA + j + (QL - 1) - i]) * SCALE_ATTN - mx);
  red[tid] = sum; __syncthreads();
  for (int s = 128; s > 0; s >>= 1) { if (tid < s) red[tid] += red[tid + s]; __syncthreads(); }
  const float inv = 1.f / red[0];
  for (int j = tid; j < KLN; j += 256) {
    float p = 0.f;
    if (j <= jmax)
      p = __expf((AC[rowA + j] + BD[rowA + j + (QL - 1) - i]) * SCALE_ATTN - mx) * inv;
    prob[rowA + j] = (_Float16)p;
  }
}

// core[i,:] = LayerNorm(core[i,:] + delta[i,:]) * g + b   (in place, D=512)
__global__ __launch_bounds__(256)
void ln_res(float* __restrict__ core, const float* __restrict__ delta,
            const float* __restrict__ g, const float* __restrict__ b) {
  __shared__ float red[256];
  const int i = blockIdx.x, tid = threadIdx.x;
  const size_t base = (size_t)i * DM;
  float x0 = core[base + tid]       + delta[base + tid];
  float x1 = core[base + 256 + tid] + delta[base + 256 + tid];
  red[tid] = x0 + x1; __syncthreads();
  for (int s = 128; s > 0; s >>= 1) { if (tid < s) red[tid] += red[tid + s]; __syncthreads(); }
  const float mu = red[0] * (1.f / 512.f); __syncthreads();
  const float d0 = x0 - mu, d1 = x1 - mu;
  red[tid] = d0 * d0 + d1 * d1; __syncthreads();
  for (int s = 128; s > 0; s >>= 1) { if (tid < s) red[tid] += red[tid + s]; __syncthreads(); }
  const float rstd = rsqrtf(red[0] * (1.f / 512.f) + 1e-5f);
  core[base + tid]       = d0 * rstd * g[tid]       + b[tid];
  core[base + 256 + tid] = d1 * rstd * g[256 + tid] + b[256 + tid];
}

// out[0,t,q,c] = y[q,c] * detr_w[t] + detr_b[t]
__global__ void post_out(const float* __restrict__ y, const float* __restrict__ dw,
                         const float* __restrict__ db, float* __restrict__ out) {
  int i = blockIdx.x * blockDim.x + threadIdx.x;
  if (i >= 4 * QL * 128) return;
  int t = i / (QL * 128);
  int r = i % (QL * 128);
  out[i] = y[r] * dw[t] + db[t];
}

// =====================================================================
// Host orchestration
// =====================================================================
static inline int gblk(long n) { long g = (n + 255) / 256; return (int)(g > 65535 ? 65535 : g); }

extern "C" void kernel_launch(void* const* d_in, const int* in_sizes, int n_in,
                              void* d_out, int out_size, void* d_ws, size_t ws_size,
                              hipStream_t stream) {
  (void)in_sizes; (void)n_in; (void)out_size; (void)ws_size;

  const float* dec_inp = (const float*)d_in[0];
  const float* mems    = (const float*)d_in[1];
  const float* enp1_w  = (const float*)d_in[2];
  const float* enp1_b  = (const float*)d_in[3];
  const float* entr_w  = (const float*)d_in[4];
  const float* entr_b  = (const float*)d_in[5];
  const float* enp2_w  = (const float*)d_in[6];
  const float* enp2_b  = (const float*)d_in[7];
  const float* r_w_b   = (const float*)d_in[8];
  const float* r_r_b   = (const float*)d_in[9];
  const float* qkv_w   = (const float*)d_in[10];
  const float* rnet_w  = (const float*)d_in[11];
  const float* o_w     = (const float*)d_in[12];
  const float* ln1_g   = (const float*)d_in[13];
  const float* ln1_b   = (const float*)d_in[14];
  const float* ff_w1   = (const float*)d_in[15];
  const float* ff_b1   = (const float*)d_in[16];
  const float* ff_w2   = (const float*)d_in[17];
  const float* ff_b2   = (const float*)d_in[18];
  const float* ln2_g   = (const float*)d_in[19];
  const float* ln2_b   = (const float*)d_in[20];
  const float* dep_w   = (const float*)d_in[21];
  const float* dep_b   = (const float*)d_in[22];
  const float* detr_w  = (const float*)d_in[23];
  const float* detr_b  = (const float*)d_in[24];

  float* out = (float*)d_out;                 // [4*1024*128] then new_mems [7*1024*512]
  float* out_mems = out + 4 * QL * 128;

  // ---- workspace bump allocator ----
  char*  ws  = (char*)d_ws;
  size_t off = 0;
  auto alloc = [&](size_t bytes) -> char* {
    off = (off + 255) & ~(size_t)255;
    char* p = ws + off;
    off += bytes;
    return p;
  };
  auto f16buf = [&](size_t n) { return (_Float16*)alloc(n * 2); };
  auto f32buf = [&](size_t n) { return (float*)alloc(n * 4); };

  // f16 weights
  _Float16* w16_enp1 = f16buf((size_t)DM * 128);
  _Float16* w16_enp2 = f16buf((size_t)DM * DM);
  _Float16* w16_qkv  = f16buf((size_t)NL * 3 * DM * DM);
  _Float16* w16_rnet = f16buf((size_t)NL * DM * DM);
  _Float16* w16_ow   = f16buf((size_t)NL * DM * DM);
  _Float16* w16_ff1  = f16buf((size_t)NL * DI * DM);
  _Float16* w16_ff2  = f16buf((size_t)NL * DM * DI);
  _Float16* w16_dep  = f16buf((size_t)128 * DM);
  // activations
  float*    bias1  = f32buf(DM);
  float*    u      = f32buf((size_t)QL * 128);
  _Float16* u16    = f16buf((size_t)QL * 128);
  float*    tmp1   = f32buf((size_t)QL * DM);
  _Float16* tmp116 = f16buf((size_t)QL * DM);
  float*    posf   = f32buf((size_t)KLN * DM);
  _Float16* pos16  = f16buf((size_t)KLN * DM);
  float*    core   = f32buf((size_t)QL * DM);
  _Float16* core16 = f16buf((size_t)QL * DM);
  _Float16* cat16  = f16buf((size_t)KLN * DM);
  float*    heads  = f32buf((size_t)KLN * 3 * DM);
  float*    rk     = f32buf((size_t)KLN * DM);
  _Float16* rk16   = f16buf((size_t)KLN * DM);
  _Float16* qw16   = f16buf((size_t)QL * DM);
  _Float16* qr16   = f16buf((size_t)QL * DM);
  _Float16* k16    = f16buf((size_t)KLN * DM);
  _Float16* v16    = f16buf((size_t)KLN * DM);
  float*    AC     = f32buf((size_t)QL * KLN);
  float*    BD     = f32buf((size_t)QL * KLN);
  _Float16* prob16 = f16buf((size_t)QL * KLN);
  float*    vec    = f32buf((size_t)QL * DM);
  _Float16* vec16  = f16buf((size_t)QL * DM);
  float*    attn   = f32buf((size_t)QL * DM);
  float*    ff1    = f32buf((size_t)QL * DI);
  _Float16* ff116  = f16buf((size_t)QL * DI);
  float*    ff2    = f32buf((size_t)QL * DM);
  float*    ybuf   = f32buf((size_t)QL * 128);

  auto cvt = [&](const float* s, _Float16* d, long n) {
    cvt_f32_to_f16<<<gblk(n), 256, 0, stream>>>(s, d, (int)n);
  };
  auto gemm = [&](const _Float16* A, int lda, const _Float16* B, int ldb, int transB,
                  float* C, int ldc, const float* bias, int M, int N, int K,
                  float alpha, int relu) {
    dim3 grid((N + BN - 1) / BN, (M + BM - 1) / BM);
    gemm_wmma_f16<<<grid, 256, 0, stream>>>(A, lda, B, ldb, transB, C, ldc, bias,
                                            M, N, K, alpha, relu);
  };

  // ---- weight conversion ----
  cvt(enp1_w, w16_enp1, (long)DM * 128);
  cvt(enp2_w, w16_enp2, (long)DM * DM);
  cvt(qkv_w,  w16_qkv,  (long)NL * 3 * DM * DM);
  cvt(rnet_w, w16_rnet, (long)NL * DM * DM);
  cvt(o_w,    w16_ow,   (long)NL * DM * DM);
  cvt(ff_w1,  w16_ff1,  (long)NL * DI * DM);
  cvt(ff_w2,  w16_ff2,  (long)NL * DM * DI);
  cvt(dep_w,  w16_dep,  (long)128 * DM);

  // ---- preprocess: word_emb -> core ----
  track_combine<<<gblk((long)QL * 128), 256, 0, stream>>>(dec_inp, entr_w, u);
  make_bias1<<<2, 256, 0, stream>>>(enp1_b, entr_w, entr_b, bias1);
  cvt(u, u16, (long)QL * 128);
  gemm(u16, 128, w16_enp1, 128, 1, tmp1, DM, bias1, QL, DM, 128, 1.f, 0);
  cvt(tmp1, tmp116, (long)QL * DM);
  gemm(tmp116, DM, w16_enp2, DM, 1, core, DM, enp2_b, QL, DM, DM, 1.f, 0);
  copy_f32<<<gblk((long)QL * DM), 256, 0, stream>>>(out_mems, core, QL * DM); // new_mems[0]

  // ---- relative positional embedding ----
  pos_emb_make<<<gblk((long)KLN * DM), 256, 0, stream>>>(posf);
  cvt(posf, pos16, (long)KLN * DM);

  // ---- layers ----
  for (int l = 0; l < NL; ++l) {
    const _Float16* qkv16 = w16_qkv  + (size_t)l * 3 * DM * DM;
    const _Float16* rn16  = w16_rnet + (size_t)l * DM * DM;
    const _Float16* ow16  = w16_ow   + (size_t)l * DM * DM;
    const _Float16* f116  = w16_ff1  + (size_t)l * DI * DM;
    const _Float16* f216  = w16_ff2  + (size_t)l * DM * DI;

    concat_cat16<<<gblk((long)KLN * DM), 256, 0, stream>>>(mems + (size_t)l * QL * DM, core, cat16);
    gemm(cat16, DM, qkv16, DM, 1, heads, 3 * DM, nullptr, KLN, 3 * DM, DM, 1.f, 0);
    gemm(pos16, DM, rn16, DM, 1, rk, DM, nullptr, KLN, DM, DM, 1.f, 0);
    cvt(rk, rk16, (long)KLN * DM);
    split_qkv<<<gblk((long)KLN * 3 * DM), 256, 0, stream>>>(heads, r_w_b, r_r_b, qw16, qr16, k16, v16);

    for (int h = 0; h < NH; ++h) {
      const int o = h * DHD;
      gemm(qw16 + o, DM, k16 + o, DM, 1, AC, KLN, nullptr, QL, KLN, DHD, 1.f, 0);
      gemm(qr16 + o, DM, rk16 + o, DM, 1, BD, KLN, nullptr, QL, KLN, DHD, 1.f, 0);
      softmax_rel<<<QL, 256, 0, stream>>>(AC, BD, prob16);
      gemm(prob16, KLN, v16 + o, DM, 0, vec + o, DM, nullptr, QL, DHD, KLN, 1.f, 0);
    }

    cvt(vec, vec16, (long)QL * DM);
    gemm(vec16, DM, ow16, DM, 1, attn, DM, nullptr, QL, DM, DM, 1.f, 0);
    ln_res<<<QL, 256, 0, stream>>>(core, attn, ln1_g + l * DM, ln1_b + l * DM);

    cvt(core, core16, (long)QL * DM);
    gemm(core16, DM, f116, DM, 1, ff1, DI, ff_b1 + l * DI, QL, DI, DM, 1.f, 1);
    cvt(ff1, ff116, (long)QL * DI);
    gemm(ff116, DI, f216, DI, 1, ff2, DM, ff_b2 + l * DM, QL, DM, DI, 1.f, 0);
    ln_res<<<QL, 256, 0, stream>>>(core, ff2, ln2_g + l * DM, ln2_b + l * DM);

    copy_f32<<<gblk((long)QL * DM), 256, 0, stream>>>(out_mems + (size_t)(1 + l) * QL * DM,
                                                      core, QL * DM);
  }

  // ---- postprocess ----
  cvt(core, core16, (long)QL * DM);
  gemm(core16, DM, w16_dep, DM, 1, ybuf, 128, dep_b, QL, 128, DM, 1.f, 0);
  post_out<<<gblk((long)4 * QL * 128), 256, 0, stream>>>(ybuf, detr_w, detr_b, out);
}
